// FavorPlusAttention_39384850104919
// MI455X (gfx1250) — matrix operand order repaired and checked by reading, not compiled, not run on hardware
//
#include <hip/hip_runtime.h>
#include <math.h>
#include <stdint.h>

// Problem constants (reference: B=4, T=4096, d=1024, h=16, m=256, dk=64)
#define TB   4
#define TT   4096
#define TD   1024
#define THH  16
#define TDK  64
#define TM   256
#define NROWS (TB*TT)     // 16384
#define BHN   (TB*THH)    // 64
#define KV_CH 32          // K-chunk (t rows) staged per TDM transfer in k_kv

typedef __attribute__((ext_vector_type(16))) _Float16 h16;
typedef __attribute__((ext_vector_type(8)))  float    f8;
typedef __attribute__((ext_vector_type(4)))  unsigned int u32x4;
typedef __attribute__((ext_vector_type(8)))  int      i32x8v;
typedef __attribute__((ext_vector_type(4)))  int      i32x4v;

static __device__ __forceinline__ f8 wmma16(h16 a, h16 b, f8 c) {
  // D = A(16x32 f16) * B(32x16 f16) + C(16x16 f32)
  return __builtin_amdgcn_wmma_f32_16x16x32_f16(false, a, false, b, (short)0, c, false, false);
}
// A-fragment K offset (vgpr v, lane-half hf): lanes0-15 hold K {0-7,16-23}, lanes16-31 {8-15,24-31}
static __device__ __forceinline__ int ka_off(int v, int hf) {
  return (hf ? 8 : 0) + ((v & 4) ? 16 : 0) + ((v & 3) << 1);
}
// B-fragment K offset: lanes0-15 hold K {0-7,16-23}, lanes16-31 {8-15,24-31}
static __device__ __forceinline__ int kb_off(int v, int hf) {
  return (hf ? 8 : 0) + ((v & 4) ? 16 : 0) + ((v & 3) << 1);
}
static __device__ __forceinline__ f8 fzero() {
  f8 z;
#pragma unroll
  for (int i = 0; i < 8; ++i) z[i] = 0.0f;
  return z;
}

// ---------------------------------------------------------------------------
// Tensor Data Mover: async 2D tile load Global -> LDS (CDNA5 TDM, D# in SGPRs).
// Descriptor packing per CDNA5 ISA ch.8 (group0 128b, group1 256b; groups 2/3
// zero for a 2D tensor). data_size code: 0=1B,1=2B,2=4B,3=8B. Dims/strides in
// elements. One wave issues it; pair with s_wait_tensorcnt + block barrier.
// 6-arg builtin form: (u32x4 g0, i32x8 g1, i32x4 g2, i32x4 g3, i32x8, i32 cpol)
// ---------------------------------------------------------------------------
static __device__ __forceinline__ void tdm_load_2d(uint32_t lds_off,
                                                   const void* gptr,
                                                   uint32_t data_size_code,
                                                   uint32_t tensor_d0, uint32_t tensor_d1,
                                                   uint32_t tile_d0, uint32_t tile_d1,
                                                   uint64_t stride0) {
  const uint64_t ga = (uint64_t)(uintptr_t)gptr;
  u32x4 g0;
  g0[0] = 1u;                                            // count=1, user descriptor
  g0[1] = lds_off;                                       // LDS byte address
  g0[2] = (uint32_t)(ga & 0xffffffffu);                  // global_addr[31:0]
  g0[3] = (uint32_t)((ga >> 32) & 0x01ffffffu) | (2u << 30);  // addr[56:32] | type=2
  i32x8v g1;
  g1[0] = (int)(data_size_code << 16);                   // wg_mask=0, data_size
  g1[1] = (int)((tensor_d0 & 0xffffu) << 16);            // tensor_dim0[15:0] @ bits63:48
  g1[2] = (int)(((tensor_d0 >> 16) & 0xffffu) | ((tensor_d1 & 0xffffu) << 16));
  g1[3] = (int)(((tensor_d1 >> 16) & 0xffffu) | ((tile_d0 & 0xffffu) << 16));
  g1[4] = (int)(tile_d1 & 0xffffu);                      // tile_dim1, tile_dim2=0
  g1[5] = (int)(uint32_t)(stride0 & 0xffffffffu);        // tensor_dim0_stride[31:0]
  g1[6] = (int)(uint32_t)((stride0 >> 32) & 0xffffu);    // stride0[47:32], stride1=0
  g1[7] = 0;
  i32x4v z4;
  z4[0] = 0; z4[1] = 0; z4[2] = 0; z4[3] = 0;
  i32x8v z8;
#pragma unroll
  for (int i = 0; i < 8; ++i) z8[i] = 0;
  __builtin_amdgcn_tensor_load_to_lds(g0, g1, z4, z4, z8, 0);
}
static __device__ __forceinline__ uint32_t lds_addr_of(const void* p) {
  return (uint32_t)(uintptr_t)p;   // low 32 bits of flat LDS pointer = LDS offset
}

// ---------------------------------------------------------------------------
// Kernel 0: f32 -> f16 bulk convert (inputs converted once; removes per-GEMM
// in-loop v_cvt work and halves operand load bytes).
// ---------------------------------------------------------------------------
__global__ __launch_bounds__(256) void k_cvt(const float* __restrict__ src,
                                             _Float16* __restrict__ dst, int n) {
  const int i = (blockIdx.x * 256 + threadIdx.x) * 4;
  if (i < n) {
#pragma unroll
    for (int j = 0; j < 4; ++j) dst[i + j] = (_Float16)src[i + j];
  }
}

// ---------------------------------------------------------------------------
// Kernel 1: C = X[16384,1024] @ W^T -> scaled f16, laid out [B,h,T,dk].
// Block: 8 waves * 16 rows = 128 rows, 64 cols. Grid (128, 16). f16 operands.
// ---------------------------------------------------------------------------
__global__ __launch_bounds__(256) void k_qkv(const _Float16* __restrict__ X,
                                             const _Float16* __restrict__ W,
                                             _Float16* __restrict__ Out,
                                             float scale) {
  const int tid = threadIdx.x, wv = tid >> 5, ln = tid & 31, lp = ln & 15, hf = ln >> 4;
  const int m0 = blockIdx.x * 128 + wv * 16;
  const int n0 = blockIdx.y * 64;
  f8 acc[4];
#pragma unroll
  for (int j = 0; j < 4; ++j) acc[j] = fzero();

  const _Float16* arow = X + (size_t)(m0 + lp) * TD;
  for (int k0 = 0; k0 < TD; k0 += 32) {
    __builtin_prefetch(arow + k0 + 256, 0, 3);   // global_prefetch_b8, WGP scope
    h16 a;
#pragma unroll
    for (int v = 0; v < 8; ++v) {
      const int k = k0 + ka_off(v, hf);
      a[2 * v]     = arow[k];
      a[2 * v + 1] = arow[k + 1];
    }
#pragma unroll
    for (int j = 0; j < 4; ++j) {
      const _Float16* wrow = W + (size_t)(n0 + 16 * j + lp) * TD;  // B[k,n] = W[n,k]
      h16 b;
#pragma unroll
      for (int v = 0; v < 8; ++v) {
        const int k = k0 + kb_off(v, hf);
        b[2 * v]     = wrow[k];
        b[2 * v + 1] = wrow[k + 1];
      }
      acc[j] = wmma16(a, b, acc[j]);
    }
  }
#pragma unroll
  for (int j = 0; j < 4; ++j) {
    const int n = n0 + 16 * j + lp;
    const int hh = n >> 6, c = n & 63;
#pragma unroll
    for (int r = 0; r < 8; ++r) {
      const int mrow = m0 + r + 8 * hf;
      const int bI = mrow / TT, t = mrow % TT;
      Out[(((size_t)bI * THH + hh) * TT + t) * TDK + c] = (_Float16)(acc[j][r] * scale);
    }
  }
}

// ---------------------------------------------------------------------------
// Kernel 2 (fused): xw = Q_bh[T,64] @ proj_h^T; phi = (exp(xw-rowmax) *
// exp(-0.5*||q||^2) + eps)/sqrt(m) -> f16. proj head tile (256x64 f16, 32KB)
// staged into LDS by the Tensor Data Mover, overlapped with the norm phase.
// One wave: 16 rows x 256 cols (16 WMMA tiles). Grid (32, 64).
// ---------------------------------------------------------------------------
__global__ __launch_bounds__(256) void k_phi(const _Float16* __restrict__ QK,
                                             const _Float16* __restrict__ Pj,
                                             _Float16* __restrict__ Phi) {
  __shared__ _Float16 sP[TM * TDK];   // 32 KB: proj head, row-major [m, dk]
  __shared__ float spart[256];
  __shared__ float sbase[128];
  const int bh = blockIdx.y;
  const int t0 = blockIdx.x * 128;
  const _Float16* A = QK + (size_t)bh * TT * TDK;
  const _Float16* P = Pj + (size_t)(bh & (THH - 1)) * TM * TDK;
  _Float16* O = Phi + (size_t)bh * TT * TM;

  const int tid = threadIdx.x, wv = tid >> 5, ln = tid & 31, lp = ln & 15, hf = ln >> 4;
  if (wv == 0) {  // one wave kicks the TDM; DMA overlaps the norm phase below
    tdm_load_2d(lds_addr_of(sP), P, /*2B*/1, TDK, TM, TDK, TM, TDK);
  }

  // exp(-0.5 * ||row||^2) for the block's 128 rows
  {
    const int i = tid, rl = i & 127, part = i >> 7;
    const _Float16* q = A + (size_t)(t0 + rl) * TDK + part * 32;
    float s = 0.0f;
#pragma unroll
    for (int jj = 0; jj < 32; ++jj) { const float qv = (float)q[jj]; s += qv * qv; }
    spart[i] = s;
  }
  __syncthreads();
  if (tid < 128)
    sbase[tid] = __expf(-0.5f * (spart[tid] + spart[tid + 128]));
  __builtin_amdgcn_s_wait_tensorcnt(0);   // issuing wave drains TDM; others pass
  __syncthreads();

  const int m0 = t0 + wv * 16;
  f8 acc[16];
#pragma unroll
  for (int jj = 0; jj < 16; ++jj) acc[jj] = fzero();

  const _Float16* arow = A + (size_t)(m0 + lp) * TDK;
  for (int k0 = 0; k0 < TDK; k0 += 32) {
    h16 a;
#pragma unroll
    for (int v = 0; v < 8; ++v) {
      const int k = k0 + ka_off(v, hf);
      a[2 * v]     = arow[k];
      a[2 * v + 1] = arow[k + 1];
    }
#pragma unroll
    for (int jj = 0; jj < 16; ++jj) {
      const _Float16* prow = sP + (size_t)(16 * jj + lp) * TDK;  // B[k,n] = proj[n,k]
      h16 b;
#pragma unroll
      for (int v = 0; v < 8; ++v) {
        const int k = k0 + kb_off(v, hf);
        b[2 * v]     = prow[k];
        b[2 * v + 1] = prow[k + 1];
      }
      acc[jj] = wmma16(a, b, acc[jj]);
    }
  }

  // Row-max over 256 cols: across 16 tiles, then across the 16 lanes of a half
  float mx[8];
#pragma unroll
  for (int r = 0; r < 8; ++r) {
    mx[r] = -1e30f;
#pragma unroll
    for (int jj = 0; jj < 16; ++jj) mx[r] = fmaxf(mx[r], acc[jj][r]);
    for (int off = 8; off >= 1; off >>= 1) mx[r] = fmaxf(mx[r], __shfl_xor(mx[r], off, 32));
  }

  const float rsm = 0.0625f;  // 1/sqrt(256)
#pragma unroll
  for (int r = 0; r < 8; ++r) {
    const int rl = wv * 16 + r + 8 * hf;
    const float base = sbase[rl];
#pragma unroll
    for (int jj = 0; jj < 16; ++jj) {
      const float val = (__expf(acc[jj][r] - mx[r]) * base + 1e-6f) * rsm;
      O[(size_t)(t0 + rl) * TM + 16 * jj + lp] = (_Float16)val;
    }
  }
}

// ---------------------------------------------------------------------------
// Kernel 3: per (b,h):  KV[m,dk] = phiK^T @ V  (M=256, N=64, K=4096) and
// Ksum[m] = phiK^T @ ones. K loop is TDM-pipelined: double-buffered 32x128
// phiK slices (8KB) + 32x64 V slices (4KB) staged to LDS; s_wait_tensorcnt 2
// overlaps the next chunk's DMA with the current chunk's 5 WMMAs.
// Grid (2, 64). Wave: 16 m-rows x 64 cols.
// ---------------------------------------------------------------------------
__global__ __launch_bounds__(256) void k_kv(const _Float16* __restrict__ PhiK,
                                            const _Float16* __restrict__ V,
                                            _Float16* __restrict__ KVm,
                                            float* __restrict__ Ksum) {
  __shared__ _Float16 sPK[2][KV_CH * 128];  // 2 x 8 KB: phiK[t-chunk][m-halfblock]
  __shared__ _Float16 sV[2][KV_CH * TDK];   // 2 x 4 KB: V[t-chunk][dk]
  const int tid = threadIdx.x, wv = tid >> 5, ln = tid & 31, lp = ln & 15, hf = ln >> 4;
  const int bh = blockIdx.y;
  const int mblock = blockIdx.x * 128;
  const _Float16* PK = PhiK + (size_t)bh * TT * TM;
  const _Float16* Vb = V + (size_t)bh * TT * TDK;

  f8 acc[4];
#pragma unroll
  for (int j = 0; j < 4; ++j) acc[j] = fzero();
  f8 accs = fzero();
  h16 ones;
#pragma unroll
  for (int e = 0; e < 16; ++e) ones[e] = (_Float16)1.0f;

  const int NCH = TT / KV_CH;  // 128 chunks
  if (wv == 0) {
    tdm_load_2d(lds_addr_of(&sPK[0][0]), PK + mblock, 1, TM, KV_CH, 128, KV_CH, TM);
    tdm_load_2d(lds_addr_of(&sV[0][0]), Vb, 1, TDK, KV_CH, TDK, KV_CH, TDK);
  }
  for (int c = 0; c < NCH; ++c) {
    const int cur = c & 1, nxt = cur ^ 1;
    if (wv == 0) {
      if (c + 1 < NCH) {
        const size_t k1 = (size_t)(c + 1) * KV_CH;
        tdm_load_2d(lds_addr_of(&sPK[nxt][0]), PK + k1 * TM + mblock, 1, TM, KV_CH, 128, KV_CH, TM);
        tdm_load_2d(lds_addr_of(&sV[nxt][0]), Vb + k1 * TDK, 1, TDK, KV_CH, TDK, KV_CH, TDK);
        __builtin_amdgcn_s_wait_tensorcnt(2);  // current chunk done; next in flight
      } else {
        __builtin_amdgcn_s_wait_tensorcnt(0);
      }
    }
    __syncthreads();
    const _Float16* pk = &sPK[cur][0];
    const _Float16* vv = &sV[cur][0];
    h16 a;  // A[m,k] = phiK[k, m]: transposed read from LDS chunk
#pragma unroll
    for (int v = 0; v < 8; ++v) {
      const int k = ka_off(v, hf);
      a[2 * v]     = pk[k * 128 + (wv * 16 + lp)];
      a[2 * v + 1] = pk[(k + 1) * 128 + (wv * 16 + lp)];
    }
#pragma unroll
    for (int j = 0; j < 4; ++j) {
      h16 b;  // B[k,n] = V[k, n] from LDS chunk
#pragma unroll
      for (int v = 0; v < 8; ++v) {
        const int k = kb_off(v, hf);
        b[2 * v]     = vv[k * TDK + 16 * j + lp];
        b[2 * v + 1] = vv[(k + 1) * TDK + 16 * j + lp];
      }
      acc[j] = wmma16(a, b, acc[j]);
    }
    accs = wmma16(a, ones, accs);
    __syncthreads();  // protect buffers before they are overwritten
  }
  const int m0 = mblock + wv * 16;
#pragma unroll
  for (int j = 0; j < 4; ++j)
#pragma unroll
    for (int r = 0; r < 8; ++r)
      KVm[((size_t)bh * TM + m0 + r + 8 * hf) * TDK + 16 * j + lp] = (_Float16)acc[j][r];
  if (lp == 0) {
#pragma unroll
    for (int r = 0; r < 8; ++r) Ksum[(size_t)bh * TM + m0 + r + 8 * hf] = accs[r];
  }
}

// ---------------------------------------------------------------------------
// Kernel 4: per (b,h): num = phiQ[T,256] @ KV[256,64]; den = phiQ @ Ksum + eps.
// KV tile (256x64 f16, 32KB) staged to LDS by TDM, overlapped with the den
// phase; writes num/den into merged [B,T,d] (f16). Grid (32, 64).
// ---------------------------------------------------------------------------
__global__ __launch_bounds__(256) void k_numden(const _Float16* __restrict__ PhiQ,
                                                const _Float16* __restrict__ KVm,
                                                const float* __restrict__ Ksum,
                                                _Float16* __restrict__ Merged) {
  __shared__ _Float16 sKV[TM * TDK];  // 32 KB
  __shared__ float spart[256];
  __shared__ float sden[128];
  const int bh = blockIdx.y;
  const int bI = bh >> 4, hh = bh & 15;
  const int t0 = blockIdx.x * 128;
  const _Float16* PQ = PhiQ + (size_t)bh * TT * TM;
  const _Float16* KB = KVm + (size_t)bh * TM * TDK;
  const float* Ks = Ksum + (size_t)bh * TM;

  const int tid = threadIdx.x, wv = tid >> 5, ln = tid & 31, lp = ln & 15, hf = ln >> 4;
  if (wv == 0) {  // TDM: stage KV (row-major [256 x 64] f16) into LDS
    tdm_load_2d(lds_addr_of(sKV), KB, /*2B*/1, TDK, TM, TDK, TM, TDK);
  }

  // Denominator for the block's 128 rows (overlaps the TDM transfer)
  {
    const int i = tid, rl = i & 127, part = i >> 7;
    const _Float16* pr = PQ + (size_t)(t0 + rl) * TM + part * 128;
    const float* ks = Ks + part * 128;
    float s = 0.0f;
    for (int jj = 0; jj < 128; ++jj) s += (float)pr[jj] * ks[jj];
    spart[i] = s;
  }
  __syncthreads();
  if (tid < 128)
    sden[tid] = spart[tid] + spart[tid + 128] + 1e-6f;
  __builtin_amdgcn_s_wait_tensorcnt(0);
  __syncthreads();

  const int m0 = t0 + wv * 16;
  f8 acc[4];
#pragma unroll
  for (int j = 0; j < 4; ++j) acc[j] = fzero();

  const _Float16* arow = PQ + (size_t)(m0 + lp) * TM;
  for (int k0 = 0; k0 < TM; k0 += 32) {
    h16 a;
#pragma unroll
    for (int v = 0; v < 8; ++v) {
      const int k = k0 + ka_off(v, hf);
      a[2 * v]     = arow[k];
      a[2 * v + 1] = arow[k + 1];
    }
#pragma unroll
    for (int j = 0; j < 4; ++j) {
      h16 b;  // KV from LDS, row-major [256 x 64]
#pragma unroll
      for (int v = 0; v < 8; ++v) {
        const int k = k0 + kb_off(v, hf);
        b[2 * v]     = sKV[(size_t)k * TDK + 16 * j + lp];
        b[2 * v + 1] = sKV[(size_t)(k + 1) * TDK + 16 * j + lp];
      }
      acc[j] = wmma16(a, b, acc[j]);
    }
  }
#pragma unroll
  for (int j = 0; j < 4; ++j)
#pragma unroll
    for (int r = 0; r < 8; ++r) {
      const int rl = wv * 16 + r + 8 * hf;
      const int t = t0 + rl;
      const float o = acc[j][r] / sden[rl];
      Merged[((size_t)bI * TT + t) * TD + hh * TDK + 16 * j + lp] = (_Float16)o;
    }
}

// ---------------------------------------------------------------------------
// Kernel 5: out = Merged[16384,1024](f16) @ Wo^T + bo  -> f32. Grid (128, 16).
// ---------------------------------------------------------------------------
__global__ __launch_bounds__(256) void k_outproj(const _Float16* __restrict__ Mg,
                                                 const _Float16* __restrict__ Wo,
                                                 const float* __restrict__ bo,
                                                 float* __restrict__ Out) {
  const int tid = threadIdx.x, wv = tid >> 5, ln = tid & 31, lp = ln & 15, hf = ln >> 4;
  const int m0 = blockIdx.x * 128 + wv * 16;
  const int n0 = blockIdx.y * 64;
  f8 acc[4];
#pragma unroll
  for (int j = 0; j < 4; ++j) acc[j] = fzero();

  const _Float16* arow = Mg + (size_t)(m0 + lp) * TD;
  for (int k0 = 0; k0 < TD; k0 += 32) {
    __builtin_prefetch(arow + k0 + 256, 0, 3);
    h16 a;
#pragma unroll
    for (int v = 0; v < 8; ++v) {
      const int k = k0 + ka_off(v, hf);
      a[2 * v]     = arow[k];
      a[2 * v + 1] = arow[k + 1];
    }
#pragma unroll
    for (int j = 0; j < 4; ++j) {
      const _Float16* wrow = Wo + (size_t)(n0 + 16 * j + lp) * TD;
      h16 b;
#pragma unroll
      for (int v = 0; v < 8; ++v) {
        const int k = k0 + kb_off(v, hf);
        b[2 * v]     = wrow[k];
        b[2 * v + 1] = wrow[k + 1];
      }
      acc[j] = wmma16(a, b, acc[j]);
    }
  }
#pragma unroll
  for (int j = 0; j < 4; ++j) {
    const int n = n0 + 16 * j + lp;
    const float bv = bo[n];
#pragma unroll
    for (int r = 0; r < 8; ++r)
      Out[(size_t)(m0 + r + 8 * hf) * TD + n] = acc[j][r] + bv;
  }
}

// ---------------------------------------------------------------------------
extern "C" void kernel_launch(void* const* d_in, const int* in_sizes, int n_in,
                              void* d_out, int out_size, void* d_ws, size_t ws_size,
                              hipStream_t stream) {
  (void)in_sizes; (void)n_in; (void)out_size; (void)ws_size;
  const float* x    = (const float*)d_in[0];
  const float* Wq   = (const float*)d_in[1];
  const float* Wk   = (const float*)d_in[2];
  const float* Wv   = (const float*)d_in[3];
  const float* Wo   = (const float*)d_in[4];
  const float* bo   = (const float*)d_in[5];
  const float* proj = (const float*)d_in[6];
  float* out = (float*)d_out;

  char* ws = (char*)d_ws;
  size_t off = 0;
  auto carve = [&](size_t bytes) -> void* {
    void* p = ws + off;
    off += (bytes + 255) & ~(size_t)255;
    return p;
  };
  _Float16* Xh   = (_Float16*)carve((size_t)NROWS * TD * 2);      //  32 MB
  _Float16* Wqh  = (_Float16*)carve((size_t)TD * TD * 2);         //   2 MB
  _Float16* Wkh  = (_Float16*)carve((size_t)TD * TD * 2);
  _Float16* Wvh  = (_Float16*)carve((size_t)TD * TD * 2);
  _Float16* Woh  = (_Float16*)carve((size_t)TD * TD * 2);
  _Float16* Pjh  = (_Float16*)carve((size_t)THH * TM * TDK * 2);  // 0.5 MB
  _Float16* Qs   = (_Float16*)carve((size_t)BHN * TT * TDK * 2);  //  32 MB
  _Float16* Kss  = (_Float16*)carve((size_t)BHN * TT * TDK * 2);  //  32 MB
  _Float16* Vb   = (_Float16*)carve((size_t)BHN * TT * TDK * 2);  //  32 MB
  _Float16* PhiQ = (_Float16*)carve((size_t)BHN * TT * TM * 2);   // 128 MB
  _Float16* PhiK = (_Float16*)carve((size_t)BHN * TT * TM * 2);   // 128 MB
  _Float16* KVm  = (_Float16*)carve((size_t)BHN * TM * TDK * 2);  //   2 MB
  float*    Ksum = (float*)   carve((size_t)BHN * TM * 4);
  _Float16* Merged = Qs;  // Qs is dead after k_phi(Q); reuse (same footprint)

  const dim3 blk(256);
  const float scale = 0.125f;  // 1/sqrt(dk)

  // One-time f32 -> f16 conversion of all GEMM operands.
  k_cvt<<<dim3(NROWS * TD / 1024), blk, 0, stream>>>(x, Xh, NROWS * TD);
  k_cvt<<<dim3(TD * TD / 1024), blk, 0, stream>>>(Wq, Wqh, TD * TD);
  k_cvt<<<dim3(TD * TD / 1024), blk, 0, stream>>>(Wk, Wkh, TD * TD);
  k_cvt<<<dim3(TD * TD / 1024), blk, 0, stream>>>(Wv, Wvh, TD * TD);
  k_cvt<<<dim3(TD * TD / 1024), blk, 0, stream>>>(Wo, Woh, TD * TD);
  k_cvt<<<dim3(THH * TM * TDK / 1024), blk, 0, stream>>>(proj, Pjh, THH * TM * TDK);

  k_qkv<<<dim3(NROWS / 128, TD / 64), blk, 0, stream>>>(Xh, Wqh, Qs, scale);
  k_qkv<<<dim3(NROWS / 128, TD / 64), blk, 0, stream>>>(Xh, Wkh, Kss, scale);
  k_qkv<<<dim3(NROWS / 128, TD / 64), blk, 0, stream>>>(Xh, Wvh, Vb, 1.0f);

  k_phi<<<dim3(TT / 128, BHN), blk, 0, stream>>>(Qs, Pjh, PhiQ);
  k_phi<<<dim3(TT / 128, BHN), blk, 0, stream>>>(Kss, Pjh, PhiK);

  k_kv<<<dim3(TM / 128, BHN), blk, 0, stream>>>(PhiK, Vb, KVm, Ksum);

  k_numden<<<dim3(TT / 128, BHN), blk, 0, stream>>>(PhiQ, KVm, Ksum, Merged);

  k_outproj<<<dim3(NROWS / 128, TD / 64), blk, 0, stream>>>(Merged, Woh, bo, out);
}
